// MAF_33629593927846
// MI455X (gfx1250) — matrix operand order repaired and checked by reading, not compiled, not run on hardware
//
#include <hip/hip_runtime.h>

typedef _Float16 v16h  __attribute__((ext_vector_type(16)));
typedef _Float16 v16ha __attribute__((ext_vector_type(16), aligned(16)));
typedef float    v8f   __attribute__((ext_vector_type(8)));

#define DDIM   64
#define HDIM   16
#define NLAYER 63
#define TILE   16
#define TPW    2    // tiles per wave (fills WMMA->VALU hazard slots)
#define WAVES  4    // waves per block (128 threads)
#define NTHR   (WAVES * 32)

// f16 fast path LDS strides (halves)
#define W1HSTR 72   // [n][k] k=0..63 (+pad)  -> 36 dwords, conflict-spread
#define W2HSTR 40   // [n][k] k=0..31 (+pad)  -> 20 dwords
// fallback f32 strides (floats)
#define W1STR  68
#define W2STR  36
#define H2STR  20

// workspace layout (bytes)
#define WS1_OFF 0
#define WS1_SZ  (NLAYER * 1024 * 2)   // f16 [l][n][64]
#define WS2_OFF (WS1_OFF + WS1_SZ)
#define WS2_SZ  (NLAYER * 512 * 2)    // f16 [l][n][32] (k=16 bias row, 17..31 zeros)
#define WS3_OFF (WS2_OFF + WS2_SZ)
#define WS3_SZ  (NLAYER * 52 * 4)     // f32 [l][52]: b1[0:16) W3^T[16:48) b3[48:50) pad
#define WS_NEED (WS3_OFF + WS3_SZ)

__device__ __forceinline__ float act_tanh(float v) {
#if __has_builtin(__builtin_amdgcn_tanhf)
    return __builtin_amdgcn_tanhf(v);
#elif __has_builtin(__builtin_amdgcn_tanh_f32)
    return __builtin_amdgcn_tanh_f32(v);
#else
    float e = __expf(2.0f * v);
    return 1.0f - 2.0f * __builtin_amdgcn_rcpf(e + 1.0f);
#endif
}

// ---- CDNA5 async global->LDS copy (per-lane LDS scatter address) ----
#if __has_builtin(__builtin_amdgcn_global_load_async_to_lds_b32)
#define HAVE_ASYNC 1
typedef __attribute__((address_space(1))) int gas_int;
typedef __attribute__((address_space(3))) int las_int;
#else
#define HAVE_ASYNC 0
#endif

__device__ __forceinline__ void async_i32(const void* g, void* l) {
#if HAVE_ASYNC
    __builtin_amdgcn_global_load_async_to_lds_b32((gas_int*)g, (las_int*)l, 0, 0);
#else
    *(int*)l = *(const int*)g;
#endif
}
__device__ __forceinline__ void wait_async() {
#if HAVE_ASYNC
#if __has_builtin(__builtin_amdgcn_s_wait_asynccnt)
    __builtin_amdgcn_s_wait_asynccnt(0);
#else
    asm volatile("s_wait_asynccnt 0x0" ::: "memory");
#endif
#endif
}

struct alignas(16) H8 { _Float16 v[8]; };

// ================= prep: pack weights into f16 fragment-order images ==========
__global__ __launch_bounds__(256)
void maf_prep(const float* __restrict__ W1, const float* __restrict__ b1,
              const float* __restrict__ W2, const float* __restrict__ b2,
              const float* __restrict__ W3, const float* __restrict__ b3,
              _Float16* __restrict__ ws1, _Float16* __restrict__ ws2,
              float* __restrict__ ws3)
{
    const int l = blockIdx.x, tid = threadIdx.x;
    #pragma unroll
    for (int j = 0; j < 4; ++j) {                       // W1: [k][n] -> [n][k]
        int e = tid + 256 * j;
        int k = e >> 4, n = e & 15;
        ws1[(size_t)l * 1024 + n * 64 + k] = (_Float16)W1[(size_t)l * 1024 + e];
    }
    #pragma unroll
    for (int j = 0; j < 2; ++j) {                       // W2 + b2 row + zero pad
        int e = tid + 256 * j;
        int n = e >> 5, h = e & 31;
        float v = 0.0f;
        if      (h < 16)  v = W2[(size_t)l * 256 + h * 16 + n];
        else if (h == 16) v = b2[l * HDIM + n];
        ws2[(size_t)l * 512 + e] = (_Float16)v;
    }
    if (tid < 52) {                                     // head block (f32)
        float v = 0.0f;
        if      (tid < 16) v = b1[l * HDIM + tid];
        else if (tid < 48) { int o = (tid - 16) >> 4, k = (tid - 16) & 15;
                             v = W3[l * HDIM * 2 + k * 2 + o]; }
        else if (tid < 50) v = b3[l * 2 + (tid - 48)];
        ws3[l * 52 + tid] = v;
    }
}

// ================= fast path: f16 weight images, zero-cvt fragments ===========
__global__ __launch_bounds__(NTHR)
void maf_kernel_ws(const float* __restrict__ x,
                   const _Float16* __restrict__ ws1, const _Float16* __restrict__ ws2,
                   const float* __restrict__ ws3, const float* __restrict__ ip,
                   float* __restrict__ out, int nB)
{
    __shared__ _Float16 wt1h[2][16 * W1HSTR];
    __shared__ _Float16 wt2h[2][16 * W2HSTR];
    __shared__ float    wb  [2][52];
    __shared__ _Float16 h1s[WAVES][TPW][16 * 16];
    __shared__ float    h2s[WAVES][TPW][16 * H2STR];
    __shared__ float    zs [WAVES][TPW * 16 * 64];

    const int tid  = threadIdx.x;
    const int lane = tid & 31;
    const int wv   = tid >> 5;
    const int hf   = lane >> 4;
    const int n    = lane & 15;
    const int base = blockIdx.x * (WAVES * TPW * TILE) + wv * (TPW * TILE);

    // ---- x A-fragments for both tiles (built once) ----
    v16h ax[TPW][2];
    #pragma unroll
    for (int t = 0; t < TPW; ++t) {
        const float* xr = x + (size_t)(base + t * TILE + n) * DDIM;
        #pragma unroll
        for (int i = 0; i < 16; ++i) {
            int k = 16 * hf + i;
            ax[t][0][i] = (_Float16)xr[k];
            ax[t][1][i] = (_Float16)xr[32 + k];
        }
    }

    // ---- async stage: layer `ls` image -> buffer `buf` (dword granularity) ----
    auto stage = [&](int ls, int buf) {
        const int* g1 = (const int*)(ws1 + (size_t)ls * 1024);
        const int* g2 = (const int*)(ws2 + (size_t)ls * 512);
        const int* g3 = (const int*)(ws3 + (size_t)ls * 52);
        int* d1 = (int*)wt1h[buf];
        int* d2 = (int*)wt2h[buf];
        int* d3 = (int*)wb[buf];
        #pragma unroll
        for (int j = 0; j < 4; ++j) {          // 512 dwords of W1 image
            int e = tid + NTHR * j;
            async_i32(&g1[e], &d1[(e >> 5) * (W1HSTR / 2) + (e & 31)]);
        }
        #pragma unroll
        for (int j = 0; j < 2; ++j) {          // 256 dwords of W2 image
            int e = tid + NTHR * j;
            async_i32(&g2[e], &d2[(e >> 4) * (W2HSTR / 2) + (e & 15)]);
        }
        if (tid < 52) async_i32(&g3[tid], &d3[tid]);
    };
    stage(0, 0);

    float asum[TPW] = {0.0f, 0.0f};
    if (hf == 1) {
        #pragma unroll
        for (int t = 0; t < TPW; ++t) {
            float x0 = x[(size_t)(base + t * TILE + n) * DDIM];
            zs[wv][(t * 16 + n) * 64 + 63] = (x0 - ip[0]) * __expf(-ip[1]);
            asum[t] = ip[1];
        }
    }
    wait_async();
    __syncthreads();

    for (int l = 0; l < NLAYER; ++l) {
        const int cb = l & 1, nb = cb ^ 1;
        if (l + 1 < NLAYER) stage(l + 1, nb);  // background, ASYNCcnt-tracked

        // ---- B-fragments: raw b128 LDS loads, no conversion ----
        v16h bw0 = *(const v16ha*)&wt1h[cb][n * W1HSTR + 16 * hf];
        v16h bw1 = *(const v16ha*)&wt1h[cb][n * W1HSTR + 32 + 16 * hf];
        v16h bw2;
        {
            H8 lo = *(const H8*)&wt2h[cb][n * W2HSTR + 8 * hf];
            H8 hi = *(const H8*)&wt2h[cb][n * W2HSTR + 16 + 8 * hf];
            #pragma unroll
            for (int i = 0; i < 8; ++i) { bw2[i] = lo.v[i]; bw2[8 + i] = hi.v[i]; }
        }

        // ---- GEMM1 both tiles (independent chains fill hazard slots) ----
        float bias1 = wb[cb][n];
        v8f c1[TPW];
        #pragma unroll
        for (int t = 0; t < TPW; ++t) {
            v8f c;
            #pragma unroll
            for (int r = 0; r < 8; ++r) c[r] = 0.0f;
            c = __builtin_amdgcn_wmma_f32_16x16x32_f16(false, ax[t][0], false, bw0, (short)0, c, false, false);
            c = __builtin_amdgcn_wmma_f32_16x16x32_f16(false, ax[t][1], false, bw1, (short)0, c, false, false);
            c1[t] = c;
        }
        #pragma unroll
        for (int t = 0; t < TPW; ++t)
            #pragma unroll
            for (int r = 0; r < 8; ++r)
                h1s[wv][t][(r + 8 * hf) * 16 + n] = (_Float16)act_tanh(c1[t][r] + bias1);

        // ---- GEMM2 both tiles; b2 via K-augmented row (a2[8]=1 -> bias row) ----
        v8f c2[TPW];
        #pragma unroll
        for (int t = 0; t < TPW; ++t) {
            v16h a2;
            H8 h = *(const H8*)(&h1s[wv][t][n * 16 + 8 * hf]);
            #pragma unroll
            for (int i = 0; i < 8; ++i)  a2[i] = h.v[i];
            a2[8] = (_Float16)1.0f;
            #pragma unroll
            for (int i = 9; i < 16; ++i) a2[i] = (_Float16)0.0f;
            v8f c;
            #pragma unroll
            for (int r = 0; r < 8; ++r) c[r] = 0.0f;
            c2[t] = __builtin_amdgcn_wmma_f32_16x16x32_f16(false, a2, false, bw2, (short)0, c, false, false);
        }
        #pragma unroll
        for (int t = 0; t < TPW; ++t)
            #pragma unroll
            for (int r = 0; r < 8; ++r)
                h2s[wv][t][(r + 8 * hf) * H2STR + n] = act_tanh(c2[t][r]);

        // ---- N=2 head: lanes 0-15 -> mu, 16-31 -> alpha; float4 LDS reads ----
        const float4* w3v = (const float4*)&wb[cb][16 + 16 * hf];
        float4 w0 = w3v[0], w1v = w3v[1], w2v = w3v[2], w3q = w3v[3];
        float accs[TPW];
        #pragma unroll
        for (int t = 0; t < TPW; ++t) {
            const float4* hr = (const float4*)&h2s[wv][t][n * H2STR];
            float4 h0 = hr[0], h1 = hr[1], h2 = hr[2], h3 = hr[3];
            float a = wb[cb][48 + hf];
            a += h0.x * w0.x + h0.y * w0.y + h0.z * w0.z + h0.w * w0.w;
            a += h1.x * w1v.x + h1.y * w1v.y + h1.z * w1v.z + h1.w * w1v.w;
            a += h2.x * w2v.x + h2.y * w2v.y + h2.z * w2v.z + h2.w * w2v.w;
            a += h3.x * w3q.x + h3.y * w3q.y + h3.z * w3q.z + h3.w * w3q.w;
            accs[t] = a;
        }
        #pragma unroll
        for (int t = 0; t < TPW; ++t) {
            float mu = __shfl(accs[t], n, 32);
            if (hf == 1) {
                float alpha = accs[t];
                asum[t] += alpha;
                float xv = x[(size_t)(base + t * TILE + n) * DDIM + (l + 1)];
                zs[wv][(t * 16 + n) * 64 + (62 - l)] = (xv - mu) * __expf(-alpha);
            }
        }

        wait_async();
        __syncthreads();
    }

    {
        float4* zo = (float4*)(out + (size_t)base * DDIM);
        const float4* zsrc = (const float4*)zs[wv];
        #pragma unroll
        for (int i = 0; i < (TPW * TILE * DDIM) / (4 * 32); ++i)
            zo[i * 32 + lane] = zsrc[i * 32 + lane];
    }
    if (hf == 1) {
        #pragma unroll
        for (int t = 0; t < TPW; ++t)
            out[(size_t)nB * DDIM + base + t * TILE + n] = -asum[t];
    }
}

// ================= fallback (no/undersized workspace): round-5 version ========
__global__ __launch_bounds__(NTHR)
void maf_kernel_fb(const float* __restrict__ x,
                   const float* __restrict__ W1, const float* __restrict__ b1,
                   const float* __restrict__ W2, const float* __restrict__ b2,
                   const float* __restrict__ W3, const float* __restrict__ b3,
                   const float* __restrict__ ip,
                   float* __restrict__ out, int nB)
{
    __shared__ float wt1f[2][16 * W1STR];
    __shared__ float wt2f[2][16 * W2STR];
    __shared__ float wb  [2][50];
    __shared__ _Float16 h1s[WAVES][TPW][16 * 16];
    __shared__ float    h2s[WAVES][TPW][16 * H2STR];
    __shared__ float    zs [WAVES][TPW * 16 * 64];

    const int tid  = threadIdx.x;
    const int lane = tid & 31;
    const int wv   = tid >> 5;
    const int hf   = lane >> 4;
    const int n    = lane & 15;
    const int base = blockIdx.x * (WAVES * TPW * TILE) + wv * (TPW * TILE);

    for (int idx = tid; idx < 16 * W2STR; idx += NTHR)
        if ((idx % W2STR) > 16) { wt2f[0][idx] = 0.0f; wt2f[1][idx] = 0.0f; }

    v16h ax[TPW][2];
    #pragma unroll
    for (int t = 0; t < TPW; ++t) {
        const float* xr = x + (size_t)(base + t * TILE + n) * DDIM;
        #pragma unroll
        for (int i = 0; i < 16; ++i) {
            int k = 16 * hf + i;
            ax[t][0][i] = (_Float16)xr[k];
            ax[t][1][i] = (_Float16)xr[32 + k];
        }
    }

    auto stage = [&](int ls, int buf) {
        const float* w1n = W1 + (size_t)ls * DDIM * HDIM;
        const float* w2n = W2 + (size_t)ls * HDIM * HDIM;
        #pragma unroll
        for (int j = 0; j < 8; ++j) {
            int e = tid + NTHR * j;
            async_i32(&w1n[e], &wt1f[buf][(e & 15) * W1STR + (e >> 4)]);
        }
        #pragma unroll
        for (int j = 0; j < 2; ++j) {
            int e = tid + NTHR * j;
            async_i32(&w2n[e], &wt2f[buf][(e & 15) * W2STR + (e >> 4)]);
        }
        if (tid < 16) async_i32(&b2[ls * HDIM + tid], &wt2f[buf][tid * W2STR + 16]);
        if (tid < 50) {
            const float* src;
            if      (tid < 16) src = &b1[ls * HDIM + tid];
            else if (tid < 48) { int o = (tid - 16) >> 4, k = (tid - 16) & 15;
                                 src = &W3[ls * HDIM * 2 + k * 2 + o]; }
            else               src = &b3[ls * 2 + (tid - 48)];
            async_i32(src, &wb[buf][tid]);
        }
    };
    stage(0, 0);

    float asum[TPW] = {0.0f, 0.0f};
    if (hf == 1) {
        #pragma unroll
        for (int t = 0; t < TPW; ++t) {
            float x0 = x[(size_t)(base + t * TILE + n) * DDIM];
            zs[wv][(t * 16 + n) * 64 + 63] = (x0 - ip[0]) * __expf(-ip[1]);
            asum[t] = ip[1];
        }
    }
    wait_async();
    __syncthreads();

    for (int l = 0; l < NLAYER; ++l) {
        const int cb = l & 1, nb = cb ^ 1;
        if (l + 1 < NLAYER) stage(l + 1, nb);

        v16h bw0, bw1, bw2;
        {
            const float4* p = (const float4*)&wt1f[cb][n * W1STR + 16 * hf];
            const float4* q = (const float4*)&wt2f[cb][n * W2STR + 8 * hf];
            #pragma unroll
            for (int j = 0; j < 4; ++j) {
                float4 a = p[j], b = p[j + 8], c = q[j + (j & 2)];
                bw0[4 * j + 0] = (_Float16)a.x; bw0[4 * j + 1] = (_Float16)a.y;
                bw0[4 * j + 2] = (_Float16)a.z; bw0[4 * j + 3] = (_Float16)a.w;
                bw1[4 * j + 0] = (_Float16)b.x; bw1[4 * j + 1] = (_Float16)b.y;
                bw1[4 * j + 2] = (_Float16)b.z; bw1[4 * j + 3] = (_Float16)b.w;
                bw2[4 * j + 0] = (_Float16)c.x; bw2[4 * j + 1] = (_Float16)c.y;
                bw2[4 * j + 2] = (_Float16)c.z; bw2[4 * j + 3] = (_Float16)c.w;
            }
        }

        float bias1 = wb[cb][n];
        v8f c1[TPW];
        #pragma unroll
        for (int t = 0; t < TPW; ++t) {
            v8f c;
            #pragma unroll
            for (int r = 0; r < 8; ++r) c[r] = 0.0f;
            c = __builtin_amdgcn_wmma_f32_16x16x32_f16(false, ax[t][0], false, bw0, (short)0, c, false, false);
            c = __builtin_amdgcn_wmma_f32_16x16x32_f16(false, ax[t][1], false, bw1, (short)0, c, false, false);
            c1[t] = c;
        }
        #pragma unroll
        for (int t = 0; t < TPW; ++t)
            #pragma unroll
            for (int r = 0; r < 8; ++r)
                h1s[wv][t][(r + 8 * hf) * 16 + n] = (_Float16)act_tanh(c1[t][r] + bias1);

        v8f c2[TPW];
        #pragma unroll
        for (int t = 0; t < TPW; ++t) {
            v16h a2;
            H8 h = *(const H8*)(&h1s[wv][t][n * 16 + 8 * hf]);
            #pragma unroll
            for (int i = 0; i < 8; ++i)  a2[i] = h.v[i];
            a2[8] = (_Float16)1.0f;
            #pragma unroll
            for (int i = 9; i < 16; ++i) a2[i] = (_Float16)0.0f;
            v8f c;
            #pragma unroll
            for (int r = 0; r < 8; ++r) c[r] = 0.0f;
            c2[t] = __builtin_amdgcn_wmma_f32_16x16x32_f16(false, a2, false, bw2, (short)0, c, false, false);
        }
        #pragma unroll
        for (int t = 0; t < TPW; ++t)
            #pragma unroll
            for (int r = 0; r < 8; ++r)
                h2s[wv][t][(r + 8 * hf) * H2STR + n] = act_tanh(c2[t][r]);

        const float4* w3v = (const float4*)&wb[cb][16 + 16 * hf];
        float4 w0 = w3v[0], w1v = w3v[1], w2v = w3v[2], w3q = w3v[3];
        float accs[TPW];
        #pragma unroll
        for (int t = 0; t < TPW; ++t) {
            const float4* hr = (const float4*)&h2s[wv][t][n * H2STR];
            float4 h0 = hr[0], h1 = hr[1], h2 = hr[2], h3 = hr[3];
            float a = wb[cb][48 + hf];
            a += h0.x * w0.x + h0.y * w0.y + h0.z * w0.z + h0.w * w0.w;
            a += h1.x * w1v.x + h1.y * w1v.y + h1.z * w1v.z + h1.w * w1v.w;
            a += h2.x * w2v.x + h2.y * w2v.y + h2.z * w2v.z + h2.w * w2v.w;
            a += h3.x * w3q.x + h3.y * w3q.y + h3.z * w3q.z + h3.w * w3q.w;
            accs[t] = a;
        }
        #pragma unroll
        for (int t = 0; t < TPW; ++t) {
            float mu = __shfl(accs[t], n, 32);
            if (hf == 1) {
                float alpha = accs[t];
                asum[t] += alpha;
                float xv = x[(size_t)(base + t * TILE + n) * DDIM + (l + 1)];
                zs[wv][(t * 16 + n) * 64 + (62 - l)] = (xv - mu) * __expf(-alpha);
            }
        }

        wait_async();
        __syncthreads();
    }

    {
        float4* zo = (float4*)(out + (size_t)base * DDIM);
        const float4* zsrc = (const float4*)zs[wv];
        #pragma unroll
        for (int i = 0; i < (TPW * TILE * DDIM) / (4 * 32); ++i)
            zo[i * 32 + lane] = zsrc[i * 32 + lane];
    }
    if (hf == 1) {
        #pragma unroll
        for (int t = 0; t < TPW; ++t)
            out[(size_t)nB * DDIM + base + t * TILE + n] = -asum[t];
    }
}

extern "C" void kernel_launch(void* const* d_in, const int* in_sizes, int n_in,
                              void* d_out, int out_size, void* d_ws, size_t ws_size,
                              hipStream_t stream) {
    const float* x  = (const float*)d_in[0];
    const float* W1 = (const float*)d_in[1];
    const float* b1 = (const float*)d_in[2];
    const float* W2 = (const float*)d_in[3];
    const float* b2 = (const float*)d_in[4];
    const float* W3 = (const float*)d_in[5];
    const float* b3 = (const float*)d_in[6];
    const float* ip = (const float*)d_in[7];
    float* out = (float*)d_out;

    int nB   = in_sizes[0] / DDIM;                 // 65536
    int grid = nB / (WAVES * TPW * TILE);          // 512 blocks of 128 threads

    if (d_ws != nullptr && ws_size >= (size_t)WS_NEED) {
        _Float16* ws1 = (_Float16*)((char*)d_ws + WS1_OFF);
        _Float16* ws2 = (_Float16*)((char*)d_ws + WS2_OFF);
        float*    ws3 = (float*)   ((char*)d_ws + WS3_OFF);
        maf_prep<<<NLAYER, 256, 0, stream>>>(W1, b1, W2, b2, W3, b3, ws1, ws2, ws3);
        maf_kernel_ws<<<grid, NTHR, 0, stream>>>(x, ws1, ws2, ws3, ip, out, nB);
    } else {
        maf_kernel_fb<<<grid, NTHR, 0, stream>>>(x, W1, b1, W2, b2, W3, b3, ip, out, nB);
    }
}
